// SelfAttentionV2_51170240364574
// MI455X (gfx1250) — compile-verified, matrix operand order checked by reading, not run on hardware
//
#include <hip/hip_runtime.h>
#include <hip/hip_bf16.h>

// ---------------------------------------------------------------------------
// Self-attention for MI455X (gfx1250, wave32, WMMA bf16 16x16x32)
//   x  : [8192,1024] f32
//   Wq/Wk/Wv : [1024,1024] f32
//   out: [8192,1024] f32
// Pipeline:
//   K1: Q = xWq, K = xWk (row-major bf16), Vt = (xWv)^T (bf16)  -> workspace
//   K2: flash attention over KV tiles, f32 accumulation, f32 output
// ---------------------------------------------------------------------------

typedef __attribute__((ext_vector_type(16))) __bf16 v16bf;
typedef __attribute__((ext_vector_type(8)))  __bf16 v8bf;
typedef __attribute__((ext_vector_type(8)))  float  v8f;

union Frag16 { v16bf v; v8bf h[2]; };

#define N_ROWS 8192
#define DMODEL 1024
#define QM 32      // query rows per workgroup (attention)
#define KT 64      // kv rows per tile (attention)

static __device__ inline v8f wmma_bf16(const Frag16& a, const Frag16& b, v8f c) {
    return __builtin_amdgcn_wmma_f32_16x16x32_bf16(
        /*neg_a=*/false, a.v, /*neg_b=*/false, b.v,
        /*c_mod=*/(short)0, c, /*reuse_a=*/false, /*reuse_b=*/false);
}

// ---------------------------------------------------------------------------
// Kernel 1: QKV projection.  blockIdx.z selects Wq/Wk/Wv.
// 256 threads = 8 waves; block tile 128x128; wave tile 32x64 (2x4 WMMA tiles).
// A fragments straight from global f32 X (k-contiguous), converted to bf16.
// B fragments from LDS-staged W^T chunk (n-major, k-contiguous).
// Epilogue stages the 128x128 bf16 tile in LDS (transposed for V) and then
// does cooperative 16B vector stores with a uniform base/stride.
// ---------------------------------------------------------------------------
__global__ __launch_bounds__(256)
void qkv_proj_kernel(const float* __restrict__ X,
                     const float* __restrict__ Wq,
                     const float* __restrict__ Wk,
                     const float* __restrict__ Wv,
                     __bf16* __restrict__ Qo,
                     __bf16* __restrict__ Ko,
                     __bf16* __restrict__ Vto) {
    const int which = blockIdx.z;
    const float* __restrict__ W = (which == 0) ? Wq : (which == 1) ? Wk : Wv;
    __bf16* __restrict__ out    = (which == 0) ? Qo : (which == 1) ? Ko : Vto;

    const int m0   = blockIdx.x * 128;
    const int n0   = blockIdx.y * 128;
    const int tid  = threadIdx.x;
    const int wave = tid >> 5;
    const int lane = tid & 31;
    const int lh   = lane & 15;
    const int ksel = (lane >> 4) * 8;     // 0 or 8: bf16 WMMA per-lane k base
    const int wRow = (wave >> 1) * 32;    // wave row offset within block tile
    const int wCol = (wave & 1) * 64;     // wave col offset within block tile

    // Union: k-loop stages W^T chunk [n:128][k:32->40]; epilogue stages the
    // 128x128 bf16 output tile [r:128][c:128->136] (both 16B-aligned rows).
    __shared__ __attribute__((aligned(16))) union {
        __bf16 w[128][40];
        __bf16 c[128][136];
    } sm;

    const v8f zero8 = {0.f, 0.f, 0.f, 0.f, 0.f, 0.f, 0.f, 0.f};
    v8f acc[2][4];
    for (int mt = 0; mt < 2; ++mt)
        for (int nt = 0; nt < 4; ++nt) acc[mt][nt] = zero8;

    for (int k0 = 0; k0 < DMODEL; k0 += 32) {
        __syncthreads();   // protect previous iteration's sm.w reads
        // Stage W[k0..k0+31][n0..n0+127] transposed into LDS as bf16.
        for (int c = tid; c < 1024; c += 256) {
            const int kk = c >> 5;           // 0..31
            const int nc = (c & 31) * 4;     // 0,4,...,124
            const float4 w4 = *(const float4*)(W + (size_t)(k0 + kk) * DMODEL + n0 + nc);
            sm.w[nc + 0][kk] = (__bf16)w4.x;
            sm.w[nc + 1][kk] = (__bf16)w4.y;
            sm.w[nc + 2][kk] = (__bf16)w4.z;
            sm.w[nc + 3][kk] = (__bf16)w4.w;
        }
        __syncthreads();

        // A fragments (X rows, f32 -> bf16)
        Frag16 a[2];
        for (int mt = 0; mt < 2; ++mt) {
            const float* xp =
                X + (size_t)(m0 + wRow + mt * 16 + lh) * DMODEL + k0 + ksel;
            const float4 x0 = *(const float4*)(xp + 0);
            const float4 x1 = *(const float4*)(xp + 4);
            const float4 x2 = *(const float4*)(xp + 16);
            const float4 x3 = *(const float4*)(xp + 20);
            float t[16] = {x0.x, x0.y, x0.z, x0.w, x1.x, x1.y, x1.z, x1.w,
                           x2.x, x2.y, x2.z, x2.w, x3.x, x3.y, x3.z, x3.w};
            for (int i = 0; i < 16; ++i) a[mt].v[i] = (__bf16)t[i];
        }

        // B fragments from LDS (column of W = row of sm.w) and WMMAs
        for (int nt = 0; nt < 4; ++nt) {
            const __bf16* wp = &sm.w[wCol + nt * 16 + lh][0];
            Frag16 b;
            b.h[0] = *(const v8bf*)(wp + ksel);
            b.h[1] = *(const v8bf*)(wp + ksel + 16);
            for (int mt = 0; mt < 2; ++mt)
                acc[mt][nt] = wmma_bf16(a[mt], b, acc[mt][nt]);
        }
    }

    // ---- Epilogue: stage bf16 tile to LDS, then vector stores ----
    __syncthreads();   // all waves done reading sm.w
    const int rOff = (lane >> 4) * 8;
    if (which < 2) {
        for (int mt = 0; mt < 2; ++mt)
            for (int nt = 0; nt < 4; ++nt)
                for (int g = 0; g < 8; ++g)
                    sm.c[wRow + mt * 16 + rOff + g][wCol + nt * 16 + lh] =
                        (__bf16)acc[mt][nt][g];
    } else {
        for (int mt = 0; mt < 2; ++mt)
            for (int nt = 0; nt < 4; ++nt)
                for (int g = 0; g < 8; ++g)
                    sm.c[wCol + nt * 16 + lh][wRow + mt * 16 + rOff + g] =
                        (__bf16)acc[mt][nt][g];
    }
    __syncthreads();

    // Cooperative 16B stores; base/stride chosen once (scalar).
    __bf16* dstBase;
    size_t stride;
    if (which < 2) { dstBase = out + (size_t)m0 * DMODEL + n0; stride = DMODEL; }
    else           { dstBase = out + (size_t)n0 * N_ROWS + m0; stride = N_ROWS; }
    for (int c = tid; c < 2048; c += 256) {       // 128 rows x 16 chunks of 8 bf16
        const int r  = c >> 4;
        const int ch = (c & 15) * 8;
        const v8bf vv = *(const v8bf*)&sm.c[r][ch];
        *(v8bf*)(dstBase + (size_t)r * stride + ch) = vv;
    }
}

// ---------------------------------------------------------------------------
// Kernel 2: flash attention.
// Block = 256 threads (8 waves), owns QM=32 query rows; streams KT=64 kv rows.
//  - S phase: wave w computes S tile (mt=w>>2, nt=w&3) over k=1024, using
//    4 independent WMMA accumulator chains + global_prefetch of next K tile.
//  - softmax: all 256 threads, 8 lanes per row, shfl_xor reductions.
//  - O phase: wave w owns output cols [w*128, w*128+128); rescale + P*Vt,
//    prefetching next tile's Vt column slices.
// ---------------------------------------------------------------------------
__global__ __launch_bounds__(256)
void flash_attn_kernel(const __bf16* __restrict__ Q,
                       const __bf16* __restrict__ K,
                       const __bf16* __restrict__ Vt,
                       float* __restrict__ O) {
    const int tid  = threadIdx.x;
    const int wave = tid >> 5;
    const int lane = tid & 31;
    const int lh   = lane & 15;
    const int ksel = (lane >> 4) * 8;
    const int rOff = (lane >> 4) * 8;

    const int qBase = blockIdx.x * QM;

    __shared__ __attribute__((aligned(16))) float  sS[QM][KT];
    __shared__ __attribute__((aligned(16))) __bf16 sP[QM][KT];
    __shared__ float sM[QM], sL[QM], sC[QM];

    if (tid < QM) { sM[tid] = -3.0e38f; sL[tid] = 0.f; }

    const int smt = wave >> 2;   // S-tile row index for this wave (0..1)
    const int snt = wave & 3;    // S-tile col index for this wave (0..3)

    const v8f zero8 = {0.f, 0.f, 0.f, 0.f, 0.f, 0.f, 0.f, 0.f};
    v8f oacc[2][8];
    for (int mt = 0; mt < 2; ++mt)
        for (int nt = 0; nt < 8; ++nt) oacc[mt][nt] = zero8;

    const float scale = 0.03125f;  // 1/sqrt(1024)
    const __bf16* __restrict__ qp = Q + (size_t)(qBase + smt * 16 + lh) * DMODEL;

    for (int kv0 = 0; kv0 < N_ROWS; kv0 += KT) {
        const int kvn = (kv0 + KT) & (N_ROWS - 1);   // next tile (wrapped, in-bounds)

        // ---- S = (Q K^T) tile: 4 independent accumulator chains ----
        v8f sacc[4] = {zero8, zero8, zero8, zero8};
        const __bf16* __restrict__ kp  = K + (size_t)(kv0 + snt * 16 + lh) * DMODEL;
        const __bf16* __restrict__ kpn = K + (size_t)(kvn + snt * 16 + lh) * DMODEL;
        for (int k0 = 0; k0 < DMODEL; k0 += 128) {
            // prefetch next KV tile's K rows (2x128B covers this 256B span)
            __builtin_prefetch(kpn + k0, 0, 0);
            __builtin_prefetch(kpn + k0 + 64, 0, 0);
#pragma unroll
            for (int j = 0; j < 4; ++j) {
                const int k = k0 + j * 32;
                Frag16 a, b;
                a.h[0] = *(const v8bf*)(qp + k + ksel);
                a.h[1] = *(const v8bf*)(qp + k + ksel + 16);
                b.h[0] = *(const v8bf*)(kp + k + ksel);
                b.h[1] = *(const v8bf*)(kp + k + ksel + 16);
                sacc[j] = wmma_bf16(a, b, sacc[j]);
            }
        }
        {
            const v8f s01 = sacc[0] + sacc[1];
            const v8f s23 = sacc[2] + sacc[3];
            const v8f st  = s01 + s23;
            const int sr = smt * 16 + rOff;
            const int sc = snt * 16 + lh;
            for (int g = 0; g < 8; ++g) sS[sr + g][sc] = st[g] * scale;
        }
        __syncthreads();  // B1: S visible

        // ---- online softmax: 8 lanes per row, wave-local reductions ----
        {
            const int row = tid >> 3;        // 0..31 (rows 4w..4w+3 per wave)
            const int seg = (tid & 7) * 8;   // col segment base
            const float4 s0 = *(const float4*)&sS[row][seg];
            const float4 s1 = *(const float4*)&sS[row][seg + 4];
            float p[8] = {s0.x, s0.y, s0.z, s0.w, s1.x, s1.y, s1.z, s1.w};
            float mx = p[0];
            for (int i = 1; i < 8; ++i) mx = fmaxf(mx, p[i]);
            for (int off = 1; off < 8; off <<= 1)
                mx = fmaxf(mx, __shfl_xor(mx, off, 32));
            const float mOld = sM[row];
            const float mNew = fmaxf(mOld, mx);
            float sum = 0.f;
            v8bf pv;
            for (int i = 0; i < 8; ++i) {
                const float e = __expf(p[i] - mNew);
                sum += e;
                pv[i] = (__bf16)e;
            }
            for (int off = 1; off < 8; off <<= 1)
                sum += __shfl_xor(sum, off, 32);
            *(v8bf*)&sP[row][seg] = pv;
            if ((tid & 7) == 0) {
                const float c = __expf(mOld - mNew);
                sM[row] = mNew;
                sL[row] = sL[row] * c + sum;
                sC[row] = c;
            }
        }
        __syncthreads();  // B2: P, C visible

        // ---- O := O * c  +  P * Vt(slice) ----
        float cv[2][8];
        for (int mt = 0; mt < 2; ++mt)
            for (int g = 0; g < 8; ++g) cv[mt][g] = sC[mt * 16 + rOff + g];
        for (int mt = 0; mt < 2; ++mt)
            for (int nt = 0; nt < 8; ++nt)
                for (int g = 0; g < 8; ++g) oacc[mt][nt][g] *= cv[mt][g];

        const int colBase = wave * 128;
        for (int ks = 0; ks < 2; ++ks) {      // kv k-steps of 32
            Frag16 pa[2];
            for (int mt = 0; mt < 2; ++mt) {
                const __bf16* pp = &sP[mt * 16 + lh][ks * 32];
                pa[mt].h[0] = *(const v8bf*)(pp + ksel);
                pa[mt].h[1] = *(const v8bf*)(pp + ksel + 16);
            }
            for (int nt = 0; nt < 8; ++nt) {
                const int col = colBase + nt * 16 + lh;
                const __bf16* vp = Vt + (size_t)col * N_ROWS + kv0 + ks * 32;
                if (ks == 0) {
                    // next tile's 64 kv values for this column = one 128B line
                    __builtin_prefetch(Vt + (size_t)col * N_ROWS + kvn, 0, 0);
                }
                Frag16 b;
                b.h[0] = *(const v8bf*)(vp + ksel);
                b.h[1] = *(const v8bf*)(vp + ksel + 16);
                for (int mt = 0; mt < 2; ++mt)
                    oacc[mt][nt] = wmma_bf16(pa[mt], b, oacc[mt][nt]);
            }
        }
        __syncthreads();  // B3: done reading sP/sC/sS before next tile
    }

    // ---- final: O /= l, store f32 ----
    for (int mt = 0; mt < 2; ++mt) {
        float inv[8];
        for (int g = 0; g < 8; ++g)
            inv[g] = 1.0f / sL[mt * 16 + rOff + g];
        for (int nt = 0; nt < 8; ++nt) {
            const int col = wave * 128 + nt * 16 + lh;
            for (int g = 0; g < 8; ++g) {
                const int row = qBase + mt * 16 + rOff + g;
                O[(size_t)row * DMODEL + col] = oacc[mt][nt][g] * inv[g];
            }
        }
    }
}

// ---------------------------------------------------------------------------
extern "C" void kernel_launch(void* const* d_in, const int* in_sizes, int n_in,
                              void* d_out, int out_size, void* d_ws, size_t ws_size,
                              hipStream_t stream) {
    const float* X  = (const float*)d_in[0];
    const float* Wq = (const float*)d_in[1];
    const float* Wk = (const float*)d_in[2];
    const float* Wv = (const float*)d_in[3];
    float* Out = (float*)d_out;

    __bf16* Qb = (__bf16*)d_ws;                       // [8192][1024] bf16
    __bf16* Kb = Qb + (size_t)N_ROWS * DMODEL;        // [8192][1024] bf16
    __bf16* Vt = Kb + (size_t)N_ROWS * DMODEL;        // [1024][8192] bf16

    dim3 gProj(N_ROWS / 128, DMODEL / 128, 3);        // (64, 8, 3)
    qkv_proj_kernel<<<gProj, 256, 0, stream>>>(X, Wq, Wk, Wv, Qb, Kb, Vt);

    dim3 gAttn(N_ROWS / QM);                          // 256 workgroups
    flash_attn_kernel<<<gAttn, 256, 0, stream>>>(Qb, Kb, Vt, Out);
}
